// MultiScaleTemporalAttention_56040733278239
// MI455X (gfx1250) — compile-verified, matrix operand order checked by reading, not run on hardware
//
#include <hip/hip_runtime.h>
#include <math.h>

// ---------------------------------------------------------------------------
// MultiScaleTemporalAttention, algebraically reduced:
//   scores[b,s] = (x[b,s]·(Wk@q[b]) + q·bk) * inv_base
//   base_ctx[b] = (Σ_s attn[b,s] x[b,s]) @ Wv + bv
//   feat GEMM (67 GFLOP) in bf16 WMMA (v_wmma_f32_16x16x32_bf16), with
//   double-buffered LDS and async global->LDS staging of the (pre-transposed,
//   pre-converted) Wp tiles.
// B=64, S=1024, H=1024, NS=8.
// ---------------------------------------------------------------------------

typedef __attribute__((ext_vector_type(16))) __bf16 v16bf;
typedef __attribute__((ext_vector_type(8)))  __bf16 v8bf;
typedef __attribute__((ext_vector_type(8)))  float  v8f;

#define Bsz 64
#define Ssz 1024
#define Hsz 1024
#define NSC 8

#if defined(__HIP_DEVICE_COMPILE__) && __has_builtin(__builtin_amdgcn_global_load_async_to_lds_b128)
#define HAS_ASYNC_LDS 1
typedef int v4i __attribute__((__vector_size__(4 * sizeof(int))));
typedef __attribute__((address_space(1))) v4i GV4i;   // global int4*
typedef __attribute__((address_space(3))) v4i LV4i;   // LDS int4*
#else
#define HAS_ASYNC_LDS 0
#endif

__device__ __forceinline__ void async_wait0() {
#if HAS_ASYNC_LDS
#if __has_builtin(__builtin_amdgcn_s_wait_asynccnt)
    __builtin_amdgcn_s_wait_asynccnt(0);
#else
    asm volatile("s_wait_asynccnt 0x0" ::: "memory");
#endif
#endif
}

// segment-sum hierarchy: level L has (256>>L) rows per batch, each = sum of
// 4<<L consecutive x rows.  Row offsets of each level inside Pall[b] (510 rows).
__device__ __constant__ int c_levOff[8] = {0,256,384,448,480,496,504,508};

__device__ __forceinline__ __bf16 f2bf(float f) {
    union { float f; unsigned u; } a; a.f = f;
    unsigned r = a.u + 0x7FFFu + ((a.u >> 16) & 1u);
    union { unsigned short s; __bf16 b; } o; o.s = (unsigned short)(r >> 16);
    return o.b;
}
__device__ __forceinline__ float bf2f(__bf16 b) {
    union { unsigned short s; __bf16 b; } i; i.b = b;
    union { unsigned u; float f; } o; o.u = ((unsigned)i.s) << 16;
    return o.f;
}

// ---------------------------------------------------------------------------
// K0: WpT[s][n][k] = bf16(Wp[s][k][n])   (transpose + convert, 32x32 tiles)
// grid = (32 ktiles, 32 ntiles, 8 scales)
// ---------------------------------------------------------------------------
__global__ void __launch_bounds__(256)
k_wpT(const float* __restrict__ Wp, __bf16* __restrict__ WpT) {
    __shared__ float tile[32][33];
    int kt = blockIdx.x, nt = blockIdx.y, s = blockIdx.z;
    int c = threadIdx.x & 31, r0 = threadIdx.x >> 5;
    const float* src = Wp + (size_t)s * Hsz * Hsz;
    for (int i = 0; i < 4; i++) {
        int r = r0 + i * 8;
        tile[r][c] = src[(size_t)(kt * 32 + r) * Hsz + nt * 32 + c];
    }
    __syncthreads();
    __bf16* dst = WpT + (size_t)s * Hsz * Hsz;
    for (int i = 0; i < 4; i++) {
        int r = r0 + i * 8;
        dst[(size_t)(nt * 32 + r) * Hsz + kt * 32 + c] = f2bf(tile[c][r]);
    }
}

// ---------------------------------------------------------------------------
// K1: q = x_last@Wq + bq ; si = softmax(x_last@Ws + bs)
// ---------------------------------------------------------------------------
__global__ void __launch_bounds__(256)
k_q_si(const float* __restrict__ x, const float* __restrict__ Wq,
       const float* __restrict__ bq, const float* __restrict__ Ws,
       const float* __restrict__ bs, float* __restrict__ q,
       float* __restrict__ si) {
    int b = blockIdx.x;
    __shared__ float xl[Hsz];
    __shared__ float lg[NSC];
    const float* xr = x + ((size_t)b * Ssz + (Ssz - 1)) * Hsz;
    for (int j = threadIdx.x; j < Hsz; j += 256) xl[j] = xr[j];
    __syncthreads();
    for (int h = threadIdx.x; h < Hsz; h += 256) {
        float acc = bq[h];
        for (int j = 0; j < Hsz; j++) acc += xl[j] * Wq[(size_t)j * Hsz + h];
        q[b * Hsz + h] = acc;
    }
    if (threadIdx.x < NSC) {
        float acc = bs[threadIdx.x];
        for (int j = 0; j < Hsz; j++) acc += xl[j] * Ws[j * NSC + threadIdx.x];
        lg[threadIdx.x] = acc;
    }
    __syncthreads();
    if (threadIdx.x == 0) {
        float mx = lg[0];
        for (int n = 1; n < NSC; n++) mx = fmaxf(mx, lg[n]);
        float e[NSC], s = 0.f;
        for (int n = 0; n < NSC; n++) { e[n] = __expf(lg[n] - mx); s += e[n]; }
        for (int n = 0; n < NSC; n++) si[b * NSC + n] = e[n] / s;
    }
}

// ---------------------------------------------------------------------------
// K2: qk[b] = Wk @ q[b] ; qbk[b] = q[b]·bk
// ---------------------------------------------------------------------------
__global__ void __launch_bounds__(256)
k_qk(const float* __restrict__ q, const float* __restrict__ Wk,
     const float* __restrict__ bk, float* __restrict__ qk,
     float* __restrict__ qbk) {
    int b = blockIdx.x;
    __shared__ float qs[Hsz];
    for (int j = threadIdx.x; j < Hsz; j += 256) qs[j] = q[b * Hsz + j];
    __syncthreads();
    for (int h = threadIdx.x; h < Hsz; h += 256) {
        const float* wr = Wk + (size_t)h * Hsz;
        float acc = 0.f;
        for (int j = 0; j < Hsz; j++) acc += wr[j] * qs[j];
        qk[b * Hsz + h] = acc;
    }
    if (threadIdx.x == 0) {
        float acc = 0.f;
        for (int j = 0; j < Hsz; j++) acc += qs[j] * bk[j];
        qbk[b] = acc;
    }
}

// ---------------------------------------------------------------------------
// K3: fused stream of x: P4 (stride-4 segment sums into Pall level 0) and
// raw scores = (x·qk + qbk) * inv_base.  grid = (256 segs, 64 b)
// ---------------------------------------------------------------------------
__global__ void __launch_bounds__(256)
k_score_p4(const float* __restrict__ x, const float* __restrict__ qk,
           const float* __restrict__ qbk, float* __restrict__ scores,
           float* __restrict__ Pall) {
    int seg = blockIdx.x, b = blockIdx.y, t = threadIdx.x;
    const float* xp = x + ((size_t)b * Ssz + (size_t)seg * 4) * Hsz;
    float4 qv = *(const float4*)(qk + b * Hsz + t * 4);
    float4 p = make_float4(0.f, 0.f, 0.f, 0.f);
    float d[4];
    for (int r = 0; r < 4; r++) {
        float4 xv = *(const float4*)(xp + (size_t)r * Hsz + t * 4);
        p.x += xv.x; p.y += xv.y; p.z += xv.z; p.w += xv.w;
        d[r] = xv.x * qv.x + xv.y * qv.y + xv.z * qv.z + xv.w * qv.w;
    }
    *(float4*)(Pall + ((size_t)b * 510 + seg) * Hsz + t * 4) = p;
    __shared__ float red[8][4];
    for (int r = 0; r < 4; r++) {
        float v = d[r];
        for (int off = 16; off; off >>= 1) v += __shfl_down(v, off, 32);
        if ((t & 31) == 0) red[t >> 5][r] = v;
    }
    __syncthreads();
    if (t < 4) {
        float s = 0.f;
        for (int w = 0; w < 8; w++) s += red[w][t];
        scores[b * Ssz + seg * 4 + t] = (s + qbk[b]) * 0.03125f; // 1/sqrt(1024)
    }
}

// ---------------------------------------------------------------------------
// K4: softmax over S per batch
// ---------------------------------------------------------------------------
__global__ void __launch_bounds__(256)
k_softmax_s(const float* __restrict__ scores, float* __restrict__ attn) {
    int b = blockIdx.x, t = threadIdx.x;
    __shared__ float buf[Ssz];
    __shared__ float red[8];
    for (int j = t; j < Ssz; j += 256) buf[j] = scores[b * Ssz + j];
    __syncthreads();
    float m = -1e30f;
    for (int j = t; j < Ssz; j += 256) m = fmaxf(m, buf[j]);
    for (int off = 16; off; off >>= 1) m = fmaxf(m, __shfl_down(m, off, 32));
    if ((t & 31) == 0) red[t >> 5] = m;
    __syncthreads();
    if (t == 0) {
        float mm = red[0];
        for (int w = 1; w < 8; w++) mm = fmaxf(mm, red[w]);
        red[0] = mm;
    }
    __syncthreads();
    m = red[0];
    __syncthreads();
    float s = 0.f;
    for (int j = t; j < Ssz; j += 256) { float e = __expf(buf[j] - m); buf[j] = e; s += e; }
    for (int off = 16; off; off >>= 1) s += __shfl_down(s, off, 32);
    __syncthreads();
    if ((t & 31) == 0) red[t >> 5] = s;
    __syncthreads();
    if (t == 0) {
        float ss = 0.f;
        for (int w = 0; w < 8; w++) ss += red[w];
        red[0] = 1.0f / ss;
    }
    __syncthreads();
    float inv = red[0];
    for (int j = t; j < Ssz; j += 256) attn[b * Ssz + j] = buf[j] * inv;
}

// ---------------------------------------------------------------------------
// K5: xa[b] = Σ_s attn[b,s] x[b,s]   grid = (4 hchunks, 64 b)
// ---------------------------------------------------------------------------
__global__ void __launch_bounds__(256)
k_xa(const float* __restrict__ x, const float* __restrict__ attn,
     float* __restrict__ xa) {
    int b = blockIdx.y;
    int h = blockIdx.x * 256 + threadIdx.x;
    __shared__ float a[Ssz];
    for (int j = threadIdx.x; j < Ssz; j += 256) a[j] = attn[b * Ssz + j];
    __syncthreads();
    const float* xp = x + (size_t)b * Ssz * Hsz + h;
    float acc = 0.f;
    for (int s = 0; s < Ssz; s++) acc += a[s] * xp[(size_t)s * Hsz];
    xa[b * Hsz + h] = acc;
}

// ---------------------------------------------------------------------------
// K5b: bc = xa @ Wv + bv
// ---------------------------------------------------------------------------
__global__ void __launch_bounds__(256)
k_bc(const float* __restrict__ xa, const float* __restrict__ Wv,
     const float* __restrict__ bv, float* __restrict__ bc) {
    int b = blockIdx.x;
    __shared__ float v[Hsz];
    for (int j = threadIdx.x; j < Hsz; j += 256) v[j] = xa[b * Hsz + j];
    __syncthreads();
    for (int h = threadIdx.x; h < Hsz; h += 256) {
        float acc = bv[h];
        for (int j = 0; j < Hsz; j++) acc += v[j] * Wv[(size_t)j * Hsz + h];
        bc[b * Hsz + h] = acc;
    }
}

// ---------------------------------------------------------------------------
// K6: hierarchy halving: level L rows = pairwise sums of level L-1
// grid = (nOut, 64 b)
// ---------------------------------------------------------------------------
__global__ void __launch_bounds__(256)
k_pool_level(float* __restrict__ Pall, int inOff, int outOff) {
    int j = blockIdx.x, b = blockIdx.y, t = threadIdx.x;
    const float* i0 = Pall + ((size_t)b * 510 + inOff + 2 * j) * Hsz;
    float4 a = *(const float4*)(i0 + t * 4);
    float4 c = *(const float4*)(i0 + Hsz + t * 4);
    float4 o = make_float4(a.x + c.x, a.y + c.y, a.z + c.z, a.w + c.w);
    *(float4*)(Pall + ((size_t)b * 510 + outOff + j) * Hsz + t * 4) = o;
}

// ---------------------------------------------------------------------------
// K7: WMMA GEMM  feat = tanh(pooled @ Wp[i] + bp[i])  in bf16
// Block tile 64(M) x 128(N), BK=32, double-buffered LDS, async B staging.
// pooled row (scale i, batch b, r) = (Pall[lev,r] + Pall[lev,r+1]) / size_i
// grid = (502 Mtiles, 8 Ntiles)
// ---------------------------------------------------------------------------
#define A_LD 40
#define B_LD 40
__global__ void __launch_bounds__(256)
k_feat_gemm(const float* __restrict__ Pall, const __bf16* __restrict__ WpT,
            const float* __restrict__ bp, __bf16* __restrict__ feat) {
    __shared__ __align__(16) __bf16 As[2][64 * A_LD];
    __shared__ __align__(16) __bf16 Bs[2][128 * B_LD];

    const int mtile = blockIdx.x, ntile = blockIdx.y;
    // locate scale: tiles per scale i = n_i = 2^(i+1)-1, total 502
    int s = 0, accT = 0;
    for (; s < 8; s++) {
        int tcnt = (2 << s) - 1;
        if (mtile < accT + tcnt) break;
        accT += tcnt;
    }
    const int n_i   = (2 << s) - 1;
    const int rtile = mtile - accT;
    const int L     = 7 - s;
    const int levO  = c_levOff[L];
    const float invSize = 1.0f / (float)(8 << L);
    const size_t featRow0 = (size_t)(accT + rtile) * 64;

    const int t  = threadIdx.x;
    const int ar = t >> 2;               // A-stage row 0..63
    const int ac = (t & 3) << 3;         // A-stage col group (8 elems)
    const int mrel = rtile * 64 + ar;    // row within this scale
    const int bb = mrel / n_i;
    const int rr = mrel - bb * n_i;
    const float* ap0 = Pall + ((size_t)bb * 510 + levO + rr) * Hsz;
    const __bf16* wptS = WpT + ((size_t)s * Hsz + (size_t)ntile * 128) * Hsz;

    const int wave = t >> 5, lane = t & 31;
    const int wm = (wave >> 2) * 32, wn = (wave & 3) * 32;
    const int half = lane >> 4, ln16 = lane & 15;

    v8f cfr[2][2];
    for (int mi = 0; mi < 2; mi++)
        for (int ni = 0; ni < 2; ni++)
            for (int e = 0; e < 8; e++) cfr[mi][ni][e] = 0.f;

    // ---- prologue: stage buffer 0 with k0 = 0
    for (int c = 0; c < 2; c++) {
        int idx = t * 2 + c;
        int n  = idx >> 2;
        int kg = (idx & 3) << 3;
        const __bf16* gsrc = wptS + (size_t)n * Hsz + kg;
        __bf16* ldst = &Bs[0][n * B_LD + kg];
#if HAS_ASYNC_LDS
        __builtin_amdgcn_global_load_async_to_lds_b128((GV4i*)gsrc, (LV4i*)ldst, 0, 0);
#else
        *(uint4*)ldst = *(const uint4*)gsrc;
#endif
    }
    {
        const float* pA = ap0 + ac;
        float4 a0 = *(const float4*)(pA);
        float4 a1 = *(const float4*)(pA + 4);
        float4 b0 = *(const float4*)(pA + Hsz);
        float4 b1 = *(const float4*)(pA + Hsz + 4);
        v8bf av;
        av[0] = f2bf((a0.x + b0.x) * invSize);
        av[1] = f2bf((a0.y + b0.y) * invSize);
        av[2] = f2bf((a0.z + b0.z) * invSize);
        av[3] = f2bf((a0.w + b0.w) * invSize);
        av[4] = f2bf((a1.x + b1.x) * invSize);
        av[5] = f2bf((a1.y + b1.y) * invSize);
        av[6] = f2bf((a1.z + b1.z) * invSize);
        av[7] = f2bf((a1.w + b1.w) * invSize);
        *(v8bf*)(&As[0][ar * A_LD + ac]) = av;
    }
    async_wait0();
    __syncthreads();

    for (int it = 0; it < 32; ++it) {
        const int cur = it & 1, nxt = cur ^ 1;
        const bool haveNext = (it < 31);
        float4 a0, a1, b0, b1;
        if (haveNext) {
            const int k1 = (it + 1) * 32;
            // issue async copy of next B tile (bf16, already transposed)
            for (int c = 0; c < 2; c++) {
                int idx = t * 2 + c;
                int n  = idx >> 2;
                int kg = (idx & 3) << 3;
                const __bf16* gsrc = wptS + (size_t)n * Hsz + k1 + kg;
                __bf16* ldst = &Bs[nxt][n * B_LD + kg];
#if HAS_ASYNC_LDS
                __builtin_amdgcn_global_load_async_to_lds_b128((GV4i*)gsrc, (LV4i*)ldst, 0, 0);
#else
                *(uint4*)ldst = *(const uint4*)gsrc;
#endif
            }
            // prefetch next A operands into registers
            const float* pA = ap0 + k1 + ac;
            a0 = *(const float4*)(pA);
            a1 = *(const float4*)(pA + 4);
            b0 = *(const float4*)(pA + Hsz);
            b1 = *(const float4*)(pA + Hsz + 4);
        }

        // ---- fragments per ISA 16-bit layout: lane half picks K {0..7,16..23}
        // (half=0) or {8..15,24..31} (half=1): two contiguous b128 LDS reads
        v16bf af[2], bfr[2];
        for (int mi = 0; mi < 2; mi++) {
            int m = wm + mi * 16 + ln16;
            v8bf lo = *(const v8bf*)(&As[cur][m * A_LD + half * 8]);
            v8bf hi = *(const v8bf*)(&As[cur][m * A_LD + 16 + half * 8]);
            af[mi] = __builtin_shufflevector(lo, hi, 0,1,2,3,4,5,6,7,
                                                     8,9,10,11,12,13,14,15);
        }
        for (int ni = 0; ni < 2; ni++) {
            int n = wn + ni * 16 + ln16;
            v8bf lo = *(const v8bf*)(&Bs[cur][n * B_LD + half * 8]);
            v8bf hi = *(const v8bf*)(&Bs[cur][n * B_LD + 16 + half * 8]);
            bfr[ni] = __builtin_shufflevector(lo, hi, 0,1,2,3,4,5,6,7,
                                                      8,9,10,11,12,13,14,15);
        }
        for (int mi = 0; mi < 2; mi++)
            for (int ni = 0; ni < 2; ni++)
                cfr[mi][ni] = __builtin_amdgcn_wmma_f32_16x16x32_bf16(
                    false, af[mi], false, bfr[ni], (short)0, cfr[mi][ni],
                    false, false);

        if (haveNext) {
            v8bf av;
            av[0] = f2bf((a0.x + b0.x) * invSize);
            av[1] = f2bf((a0.y + b0.y) * invSize);
            av[2] = f2bf((a0.z + b0.z) * invSize);
            av[3] = f2bf((a0.w + b0.w) * invSize);
            av[4] = f2bf((a1.x + b1.x) * invSize);
            av[5] = f2bf((a1.y + b1.y) * invSize);
            av[6] = f2bf((a1.z + b1.z) * invSize);
            av[7] = f2bf((a1.w + b1.w) * invSize);
            *(v8bf*)(&As[nxt][ar * A_LD + ac]) = av;
            async_wait0();
        }
        __syncthreads();
    }

    // ---- epilogue: +bp, tanh, store bf16.  C/D layout: elem r -> row
    // wm+mi*16+half*8+r, col wn+ni*16+ln16
    for (int mi = 0; mi < 2; mi++)
        for (int ni = 0; ni < 2; ni++) {
            int ncol = ntile * 128 + wn + ni * 16 + ln16;
            float bpv = bp[s * Hsz + ncol];
            for (int r = 0; r < 8; r++) {
                int mrow = wm + mi * 16 + half * 8 + r;
                float v = cfr[mi][ni][r] + bpv;
                feat[(featRow0 + mrow) * Hsz + ncol] = f2bf(tanhf(v));
            }
        }
}

// ---------------------------------------------------------------------------
// K8: per-batch: loop scales, fa = softmax(q·feat/32), combined += si*Σ fa·feat
// then pre = bc + combined.  One block per batch (no atomics => deterministic)
// ---------------------------------------------------------------------------
__global__ void __launch_bounds__(256)
k_combine(const float* __restrict__ q, const __bf16* __restrict__ feat,
          const float* __restrict__ si, const float* __restrict__ bc,
          float* __restrict__ pre) {
    int b = blockIdx.x, t = threadIdx.x;
    __shared__ float qs[Hsz];
    __shared__ float fa[256];
    __shared__ float red[8];
    for (int j = t; j < Hsz; j += 256) qs[j] = q[b * Hsz + j];
    float comb[4] = {0.f, 0.f, 0.f, 0.f};
    int rowBase = 0;
    for (int sc = 0; sc < NSC; sc++) {
        int n_i = (2 << sc) - 1;
        __syncthreads();
        const __bf16* fb = feat + ((size_t)rowBase + (size_t)b * n_i) * Hsz;
        float dv = -1e30f;
        if (t < n_i) {
            const __bf16* fr = fb + (size_t)t * Hsz;
            float acc = 0.f;
            for (int j = 0; j < Hsz; j++) acc += qs[j] * bf2f(fr[j]);
            dv = acc * 0.03125f;
        }
        float m = dv;
        for (int off = 16; off; off >>= 1) m = fmaxf(m, __shfl_down(m, off, 32));
        if ((t & 31) == 0) red[t >> 5] = m;
        __syncthreads();
        if (t == 0) {
            float mm = red[0];
            for (int w = 1; w < 8; w++) mm = fmaxf(mm, red[w]);
            red[0] = mm;
        }
        __syncthreads();
        m = red[0];
        float e = (t < n_i) ? __expf(dv - m) : 0.f;
        float ssum = e;
        for (int off = 16; off; off >>= 1) ssum += __shfl_down(ssum, off, 32);
        __syncthreads();
        if ((t & 31) == 0) red[t >> 5] = ssum;
        __syncthreads();
        if (t == 0) {
            float ss = 0.f;
            for (int w = 0; w < 8; w++) ss += red[w];
            red[0] = 1.0f / ss;
        }
        __syncthreads();
        fa[t] = e * red[0];
        __syncthreads();
        float wgt = si[b * NSC + sc];
        for (int c = 0; c < 4; c++) {
            int h = t + c * 256;
            float acc = 0.f;
            for (int n = 0; n < n_i; n++) acc += fa[n] * bf2f(fb[(size_t)n * Hsz + h]);
            comb[c] += wgt * acc;
        }
        rowBase += 64 * n_i;
    }
    for (int c = 0; c < 4; c++) {
        int h = t + c * 256;
        pre[b * Hsz + h] = bc[b * Hsz + h] + comb[c];
    }
}

// ---------------------------------------------------------------------------
// K9: out = pre @ Wo + bo
// ---------------------------------------------------------------------------
__global__ void __launch_bounds__(256)
k_out(const float* __restrict__ pre, const float* __restrict__ Wo,
      const float* __restrict__ bo, float* __restrict__ out) {
    int b = blockIdx.x;
    __shared__ float v[Hsz];
    for (int j = threadIdx.x; j < Hsz; j += 256) v[j] = pre[b * Hsz + j];
    __syncthreads();
    for (int h = threadIdx.x; h < Hsz; h += 256) {
        float acc = bo[h];
        for (int j = 0; j < Hsz; j++) acc += v[j] * Wo[(size_t)j * Hsz + h];
        out[b * Hsz + h] = acc;
    }
}

// ---------------------------------------------------------------------------
extern "C" void kernel_launch(void* const* d_in, const int* in_sizes, int n_in,
                              void* d_out, int out_size, void* d_ws,
                              size_t ws_size, hipStream_t stream) {
    (void)in_sizes; (void)n_in; (void)out_size; (void)ws_size;
    const float* x  = (const float*)d_in[0];
    const float* Wq = (const float*)d_in[1];
    const float* bq = (const float*)d_in[2];
    const float* Wk = (const float*)d_in[3];
    const float* bk = (const float*)d_in[4];
    const float* Wv = (const float*)d_in[5];
    const float* bv = (const float*)d_in[6];
    const float* Wp = (const float*)d_in[7];
    const float* bp = (const float*)d_in[8];
    const float* Ws = (const float*)d_in[9];
    const float* bs = (const float*)d_in[10];
    const float* Wo = (const float*)d_in[11];
    const float* bo = (const float*)d_in[12];
    float* out = (float*)d_out;

    char* ws = (char*)d_ws;
    float* q      = (float*)(ws);
    float* qk     = (float*)(ws + (256u << 10));
    float* qbk    = (float*)(ws + (512u << 10));
    float* si     = (float*)(ws + (516u << 10));
    float* scores = (float*)(ws + (520u << 10));
    float* attn   = (float*)(ws + (776u << 10));
    float* xa     = (float*)(ws + (1032u << 10));
    float* bc     = (float*)(ws + (1288u << 10));
    float* pre    = (float*)(ws + (1544u << 10));
    float* Pall   = (float*)(ws + (2048u << 10));                  // 64*510*1024 f32
    __bf16* feat  = (__bf16*)(ws + (2048u << 10)
                               + (size_t)64 * 510 * 1024 * 4);     // 32128*1024 bf16
    __bf16* WpT   = (__bf16*)(ws + (2048u << 10)
                               + (size_t)64 * 510 * 1024 * 4
                               + (size_t)32128 * 1024 * 2);        // 8*1024*1024 bf16

    k_wpT<<<dim3(32, 32, 8), 256, 0, stream>>>(Wp, WpT);
    k_q_si<<<64, 256, 0, stream>>>(x, Wq, bq, Ws, bs, q, si);
    k_qk<<<64, 256, 0, stream>>>(q, Wk, bk, qk, qbk);
    k_score_p4<<<dim3(256, 64), 256, 0, stream>>>(x, qk, qbk, scores, Pall);
    k_softmax_s<<<64, 256, 0, stream>>>(scores, attn);
    k_xa<<<dim3(4, 64), 256, 0, stream>>>(x, attn, xa);
    k_bc<<<64, 256, 0, stream>>>(xa, Wv, bv, bc);

    static const int levOffH[8] = {0, 256, 384, 448, 480, 496, 504, 508};
    for (int L = 1; L < 8; L++) {
        int nOut = 256 >> L;
        k_pool_level<<<dim3(nOut, 64), 256, 0, stream>>>(Pall, levOffH[L - 1],
                                                         levOffH[L]);
    }

    k_feat_gemm<<<dim3(502, 8), 256, 0, stream>>>(Pall, WpT, bp, feat);
    k_combine<<<64, 256, 0, stream>>>(q, feat, si, bc, pre);
    k_out<<<64, 256, 0, stream>>>(pre, Wo, bo, out);
}
